// GNNFwd_37220186587357
// MI455X (gfx1250) — matrix operand order repaired and checked by reading, not compiled, not run on hardware
//
#include <hip/hip_runtime.h>
#include <hip/hip_bf16.h>
#include <math.h>

// ---------------------------------------------------------------------------
// MI455X / gfx1250 (CDNA5, wave32). Bipartite GNN layer:
//   BN -> bidirectional edge scatter-mean (L2-atomic bound) ->
//   WMMA GEMM (bf16 hi/lo split, f32 accumulate) + bias + residual + ReLU.
// ---------------------------------------------------------------------------

#define FEAT 64
#define WAVE 32

typedef __attribute__((ext_vector_type(16))) __bf16 v16bf;
typedef __attribute__((ext_vector_type(8)))  float  v8f;

// ---------------- zero scratch ---------------------------------------------
__global__ void zero_f32_kernel(float* p, size_t n) {
  size_t i = (size_t)blockIdx.x * blockDim.x + threadIdx.x;
  size_t stride = (size_t)gridDim.x * blockDim.x;
  for (; i < n; i += stride) p[i] = 0.0f;
}

// ---------------- per-column sum / sumsq -----------------------------------
__global__ void col_stats_kernel(const float* __restrict__ xin,
                                 float* __restrict__ sum_out,
                                 float* __restrict__ sq_out, int nrows) {
  __shared__ float s_sum[4][FEAT];
  __shared__ float s_sq[4][FEAT];
  int col = threadIdx.x & (FEAT - 1);
  int q   = threadIdx.x >> 6;                 // 4 row-lanes per block
  float s = 0.0f, sq = 0.0f;
  for (int r = blockIdx.x * 4 + q; r < nrows; r += gridDim.x * 4) {
    float v = xin[(size_t)r * FEAT + col];
    s += v; sq += v * v;
  }
  s_sum[q][col] = s;  s_sq[q][col] = sq;
  __syncthreads();
  if (threadIdx.x < FEAT) {
    float ts = s_sum[0][col] + s_sum[1][col] + s_sum[2][col] + s_sum[3][col];
    float tq = s_sq[0][col]  + s_sq[1][col]  + s_sq[2][col]  + s_sq[3][col];
    atomicAdd(sum_out + col, ts);
    atomicAdd(sq_out + col, tq);
  }
}

// ---------------- BN coefficients: scale = g*rsqrt(var+eps), shift ---------
__global__ void bn_finalize_kernel(const float* __restrict__ stats,
                                   const float* __restrict__ g_n,
                                   const float* __restrict__ b_n,
                                   const float* __restrict__ g_c,
                                   const float* __restrict__ b_c,
                                   float* __restrict__ coefs,
                                   float n_var, float n_cstr) {
  int t = threadIdx.x;
  if (t >= 2 * FEAT) return;
  int col = t & (FEAT - 1);
  bool isc = t >= FEAT;
  float N = isc ? n_cstr : n_var;
  const float* S = stats + (isc ? 2 * FEAT : 0);
  float mean = S[col] / N;
  float var  = S[FEAT + col] / N - mean * mean;
  float rs   = rsqrtf(var + 1e-5f);
  float gamma = (isc ? g_c : g_n)[col];
  float beta  = (isc ? b_c : b_n)[col];
  float sc = rs * gamma;
  float* C = coefs + (isc ? 2 * FEAT : 0);
  C[col]        = sc;
  C[FEAT + col] = beta - mean * sc;
}

// ---------------- BN apply --------------------------------------------------
__global__ void bn_apply_kernel(const float* __restrict__ xin,
                                const float* __restrict__ coef,  // [scale|shift]
                                float* __restrict__ xout, size_t n) {
  size_t i = (size_t)blockIdx.x * blockDim.x + threadIdx.x;
  size_t stride = (size_t)gridDim.x * blockDim.x;
  for (; i < n; i += stride) {
    int col = (int)(i & (FEAT - 1));
    xout[i] = fmaf(xin[i], coef[col], coef[FEAT + col]);
  }
}

// ---------------- weight prep: Wcat = [Wrel^T ; Wroot^T] (128x64) ----------
// Pre-permuted into WMMA B-matrix (32x16 bf16) lane layout so the GEMM kernel
// does one contiguous 32B v16bf load per lane per (kt,nt) fragment.
// B k order used here (16-bit, 32x16, same as A): lane = {half:1, col:4}; element e -> K = half*16+e.
// idx = ((kt*4+nt)*32 + lane)*16 + e
__global__ void prep_w_kernel(const float* __restrict__ rel,
                              const float* __restrict__ root,
                              __bf16* __restrict__ whi,
                              __bf16* __restrict__ wlo) {
  const int total = 4 * 4 * 32 * 16;  // 8192
  for (int idx = blockIdx.x * blockDim.x + threadIdx.x; idx < total;
       idx += gridDim.x * blockDim.x) {
    int e    = idx & 15;
    int lane = (idx >> 4) & 31;
    int nt   = (idx >> 9) & 3;
    int kt   = idx >> 11;
    int half = lane >> 4;
    int n    = nt * 16 + (lane & 15);
    int K    = kt * 32 + half * 16 + e;
    // out[:,n] = sum_K A[:,K] * Wcat[K][n];  Wcat[K][n] = K<64 ? Wrel[n,K] : Wroot[n,K-64]
    float w = (K < FEAT) ? rel[n * FEAT + K] : root[n * FEAT + (K - FEAT)];
    __bf16 h = (__bf16)w;
    whi[idx] = h;
    wlo[idx] = (__bf16)(w - (float)h);
  }
}

// ---------------- bidirectional edge scatter (L2-atomic bound) -------------
// One wave32 per edge; lane owns features [2l, 2l+1].
__global__ void edge_scatter_kernel(const int* __restrict__ src,
                                    const int* __restrict__ dst,
                                    const float* __restrict__ ea,
                                    const float* __restrict__ xc,
                                    const float* __restrict__ xn,
                                    float* __restrict__ nssum,
                                    float* __restrict__ ncnt,
                                    float* __restrict__ cssum,
                                    float* __restrict__ ccnt, int E) {
  int lane   = threadIdx.x & (WAVE - 1);
  int wave   = blockIdx.x * (blockDim.x >> 5) + (threadIdx.x >> 5);
  int nwaves = gridDim.x * (blockDim.x >> 5);
  for (int e = wave; e < E; e += nwaves) {
    int s = src[e];
    int d = dst[e];
    float a = ea[e];
    float2 gc = *(const float2*)(xc + (size_t)s * FEAT + 2 * lane);
    float2 gn = *(const float2*)(xn + (size_t)d * FEAT + 2 * lane);
    float* np = nssum + (size_t)d * FEAT + 2 * lane;
    atomicAdd(np,     gc.x * a);
    atomicAdd(np + 1, gc.y * a);
    float* cp = cssum + (size_t)s * FEAT + 2 * lane;
    atomicAdd(cp,     gn.x * a);
    atomicAdd(cp + 1, gn.y * a);
    if (lane == 0) {
      atomicAdd(ncnt + d, 1.0f);
      atomicAdd(ccnt + s, 1.0f);
    }
  }
}

// ---------------- WMMA GEMM + bias + residual + ReLU -----------------------
// One wave per 16-row tile.  A = [agg | x] (16x128), W = Wcat (128x64).
// bf16 hi/lo split: A*W ~= Ah*Wh + Ah*Wl + Al*Wh  (f32 accumulate).
// A layout (16-bit, 16x32): lane = {half:1, row:4};
//   element e -> localK = half*16+e  (contiguous order used here, same as B).
__global__ __launch_bounds__(256)
void gemm_out_kernel(const float* __restrict__ ssum,
                     const float* __restrict__ cntv,
                     const float* __restrict__ x,
                     const __bf16* __restrict__ wphi,
                     const __bf16* __restrict__ wplo,
                     const float* __restrict__ bias,
                     float* __restrict__ out, int nrows, int ntiles) {
  int lane = threadIdx.x & (WAVE - 1);
  int tile = blockIdx.x * (blockDim.x >> 5) + (threadIdx.x >> 5);
  if (tile >= ntiles) return;              // wave-uniform: EXEC stays all-1s

  int half = lane >> 4;
  int mr   = lane & 15;
  int row  = tile * 16 + mr;
  if (row >= nrows) row = nrows - 1;       // clamp (harmless redundant loads)

  const float* srow = ssum + (size_t)row * FEAT;
  const float* xrow = x    + (size_t)row * FEAT;
  float inv = 1.0f / fmaxf(cntv[row], 1.0f);

  const v16bf* BH = (const v16bf*)wphi;
  const v16bf* BL = (const v16bf*)wplo;

  int colbase = lane & 15;
  v8f acc[4];
#pragma unroll
  for (int nt = 0; nt < 4; ++nt) {
    float bv = bias[nt * 16 + colbase];
#pragma unroll
    for (int r = 0; r < 8; ++r) acc[nt][r] = bv;
  }

#pragma unroll
  for (int kt = 0; kt < 4; ++kt) {
    const float* base = (kt < 2) ? srow : xrow;   // agg columns then x columns
    float scale = (kt < 2) ? inv : 1.0f;
    int koff = (kt & 1) * 32;
    v16bf ahi, alo;
#pragma unroll
    for (int p = 0; p < 8; ++p) {                 // 8 contiguous float2 pairs
      int lk = half * 16 + 2 * p;
      float2 v = *(const float2*)(base + koff + lk);
      float f0 = v.x * scale, f1 = v.y * scale;
      __bf16 h0 = (__bf16)f0, h1 = (__bf16)f1;
      ahi[2 * p]     = h0;
      ahi[2 * p + 1] = h1;
      alo[2 * p]     = (__bf16)(f0 - (float)h0);
      alo[2 * p + 1] = (__bf16)(f1 - (float)h1);
    }
#pragma unroll
    for (int nt = 0; nt < 4; ++nt) {
      v16bf bhi = BH[(kt * 4 + nt) * 32 + lane];  // coalesced 32B per lane
      v16bf blo = BL[(kt * 4 + nt) * 32 + lane];
      acc[nt] = __builtin_amdgcn_wmma_f32_16x16x32_bf16(
          false, ahi, false, bhi, (short)0, acc[nt], false, false);
      acc[nt] = __builtin_amdgcn_wmma_f32_16x16x32_bf16(
          false, ahi, false, blo, (short)0, acc[nt], false, false);
      acc[nt] = __builtin_amdgcn_wmma_f32_16x16x32_bf16(
          false, alo, false, bhi, (short)0, acc[nt], false, false);
    }
  }

  // C/D layout: element r at lane -> D[r + 8*half][colbase]
#pragma unroll
  for (int nt = 0; nt < 4; ++nt) {
    int n = nt * 16 + colbase;
#pragma unroll
    for (int r = 0; r < 8; ++r) {
      int orow = tile * 16 + r + 8 * half;
      if (orow < nrows) {
        float res = x[(size_t)orow * FEAT + n];   // residual (normalized x)
        out[(size_t)orow * FEAT + n] = fmaxf(acc[nt][r] + res, 0.0f);
      }
    }
  }
}

// ---------------------------------------------------------------------------
extern "C" void kernel_launch(void* const* d_in, const int* in_sizes, int n_in,
                              void* d_out, int out_size, void* d_ws, size_t ws_size,
                              hipStream_t stream) {
  const float* var_feats  = (const float*)d_in[0];
  const float* cstr_feats = (const float*)d_in[1];
  const int*   edge_src   = (const int*)d_in[2];
  const int*   edge_dst   = (const int*)d_in[3];
  const float* edge_attr  = (const float*)d_in[4];
  const float* n_gamma    = (const float*)d_in[5];
  const float* n_beta     = (const float*)d_in[6];
  const float* c_gamma    = (const float*)d_in[7];
  const float* c_beta     = (const float*)d_in[8];
  const float* n_rel_w    = (const float*)d_in[9];
  const float* n_rel_b    = (const float*)d_in[10];
  const float* n_root_w   = (const float*)d_in[11];
  const float* c_rel_w    = (const float*)d_in[12];
  const float* c_rel_b    = (const float*)d_in[13];
  const float* c_root_w   = (const float*)d_in[14];

  const size_t nvar  = (size_t)in_sizes[0] / FEAT;   // 100000
  const size_t ncstr = (size_t)in_sizes[1] / FEAT;   // 50000
  const int    E     = in_sizes[2];                  // 1000000

  // ---- workspace layout (float units, 32B-aligned blocks) ----
  float* fws = (float*)d_ws;
  size_t off = 0;
  auto take = [&](size_t n) { float* p = fws + off; off += (n + 7) & ~(size_t)7; return p; };
  float* stats = take(4 * FEAT);            // [nsum|nsq|csum|csq]
  float* coefs = take(4 * FEAT);            // [nscale|nshift|cscale|cshift]
  float* nssum = take(nvar * FEAT);
  float* ncnt  = take(nvar);
  float* cssum = take(ncstr * FEAT);
  float* ccnt  = take(ncstr);
  size_t zero_n = off;                      // everything above is accumulated
  float* xn = take(nvar * FEAT);
  float* xc = take(ncstr * FEAT);
  __bf16* wn_hi = (__bf16*)take(4096);      // 8192 bf16 each
  __bf16* wn_lo = (__bf16*)take(4096);
  __bf16* wc_hi = (__bf16*)take(4096);
  __bf16* wc_lo = (__bf16*)take(4096);
  (void)ws_size; (void)n_in; (void)out_size;

  // 1) zero accumulators (harness does not re-poison between replays)
  zero_f32_kernel<<<4096, 256, 0, stream>>>(fws, zero_n);

  // 2) per-column stats
  col_stats_kernel<<<512, 256, 0, stream>>>(var_feats,  stats,            stats + FEAT,     (int)nvar);
  col_stats_kernel<<<512, 256, 0, stream>>>(cstr_feats, stats + 2 * FEAT, stats + 3 * FEAT, (int)ncstr);

  // 3) BN coefficients
  bn_finalize_kernel<<<1, 128, 0, stream>>>(stats, n_gamma, n_beta, c_gamma, c_beta,
                                            coefs, (float)nvar, (float)ncstr);

  // 4) BN apply
  bn_apply_kernel<<<4096, 256, 0, stream>>>(var_feats,  coefs,            xn, nvar * FEAT);
  bn_apply_kernel<<<2048, 256, 0, stream>>>(cstr_feats, coefs + 2 * FEAT, xc, ncstr * FEAT);

  // 5) weight prep (bf16 hi/lo, WMMA B-layout)
  prep_w_kernel<<<32, 256, 0, stream>>>(n_rel_w, n_root_w, wn_hi, wn_lo);
  prep_w_kernel<<<32, 256, 0, stream>>>(c_rel_w, c_root_w, wc_hi, wc_lo);

  // 6) bidirectional edge scatter (one wave per edge)
  edge_scatter_kernel<<<8192, 256, 0, stream>>>(edge_src, edge_dst, edge_attr,
                                                xc, xn, nssum, ncnt, cssum, ccnt, E);

  // 7) WMMA GEMM + bias + residual + ReLU -> concatenated output
  float* out_node = (float*)d_out;
  float* out_cstr = out_node + nvar * FEAT;
  int tiles_n = (int)((nvar  + 15) / 16);
  int tiles_c = (int)((ncstr + 15) / 16);
  gemm_out_kernel<<<(tiles_n + 7) / 8, 256, 0, stream>>>(
      nssum, ncnt, xn, wn_hi, wn_lo, n_rel_b, out_node, (int)nvar, tiles_n);
  gemm_out_kernel<<<(tiles_c + 7) / 8, 256, 0, stream>>>(
      cssum, ccnt, xc, wc_hi, wc_lo, c_rel_b, out_cstr, (int)ncstr, tiles_c);
}
